// CausalSelfAttention_75582834475065
// MI455X (gfx1250) — matrix operand order repaired and checked by reading, not compiled, not run on hardware
//
#include <hip/hip_runtime.h>
#include <hip/hip_bf16.h>
#include <stdint.h>

// ---------------------------------------------------------------------------
// Sliding-window causal attention, bf16 WMMA pipeline for gfx1250 (wave32).
// B=4, T=2048, C=1024, H=16, D=64, WINDOW=256.
// GEMMs: double-buffered LDS tiles filled with GLOBAL_LOAD_ASYNC_TO_LDS_B128
// (ASYNCcnt + s_wait_asynccnt), 16x16x32 bf16 WMMA, fp32 accumulate.
// K-loop kept at unroll 1 so accumulators stay pinned in their VGPRs
// (avoids v_mov_b64 copies + WMMA->VALU hazard nops seen with unroll 2).
// ---------------------------------------------------------------------------

typedef __bf16 bf16;
typedef __attribute__((ext_vector_type(16))) __bf16 v16bf;
typedef __attribute__((ext_vector_type(8)))  __bf16 v8bf;
typedef __attribute__((ext_vector_type(8)))  float  v8f;

#define WMMA_BF16(A, B, C) \
  __builtin_amdgcn_wmma_f32_16x16x32_bf16(false, (A), false, (B), (short)0, (C), false, false)

static __device__ inline v16bf join16(v8bf lo, v8bf hi) {
  v16bf r;
#pragma unroll
  for (int i = 0; i < 8; ++i) { r[i] = lo[i]; r[i + 8] = hi[i]; }
  return r;
}

static __device__ inline v8f vzero8() {
  v8f z;
#pragma unroll
  for (int i = 0; i < 8; ++i) z[i] = 0.0f;
  return z;
}

// A-fragment (16x32 bf16), contiguous k order (matches B):
// lanes 0-15 (g=0): K = k+0..15 (elems 0-15)
// lanes 16-31 (g=1): K = k+16..31
static __device__ inline v16bf load_a_g(const bf16* __restrict__ base, int ld, int row, int k, int g) {
  const bf16* p = base + (size_t)row * ld + k + 16 * g;
  return join16(*(const v8bf*)p, *(const v8bf*)(p + 8));
}
// B-fragment (32x16 bf16): one contiguous 32B chunk per lane.
static __device__ inline v16bf load_b_g(const bf16* __restrict__ p) {
  return join16(*(const v8bf*)p, *(const v8bf*)(p + 8));
}

// ---------------------------------------------------------------------------
// Async global->LDS (gfx1250).  VDST = 32-bit LDS byte offset (low 32 bits of
// the generic pointer), VADDR = 64-bit global address.  Tracked by ASYNCcnt.
// ---------------------------------------------------------------------------
static __device__ inline void async_b128(const bf16* lds_ptr, const bf16* g_ptr) {
  unsigned loff = (unsigned)(uintptr_t)lds_ptr;
  asm volatile("global_load_async_to_lds_b128 %0, %1, off"
               :: "v"(loff), "v"(g_ptr) : "memory");
}
static __device__ inline void wait_async_le4() {
  asm volatile("s_wait_asynccnt 4" ::: "memory");
}
static __device__ inline void wait_async_0() {
  asm volatile("s_wait_asynccnt 0" ::: "memory");
}

// ---------------------------------------------------------------------------
// Shared GEMM core: 128x128 block tile, K=1024, k-step 32.
// LDS tiles stored [row][LDH=40 halves] (80B stride -> conflict-free frags).
// Each of 256 threads async-copies 32B of A and 32B of B per k-step.
// ---------------------------------------------------------------------------
#define LDH 40          // halves per LDS row (80 bytes)
#define TILE_HALVES (128 * LDH)

static __device__ inline void issue_tiles(const bf16* __restrict__ Xg,
                                          const bf16* __restrict__ Wg,
                                          int m0b, int n0b, int k,
                                          bf16* sA, bf16* sB, int tid) {
  const int row = tid >> 1, half = tid & 1;
  const bf16* ga = Xg + (size_t)(m0b + row) * 1024 + k + 16 * half;
  const bf16* gb = Wg + (size_t)(n0b + row) * 1024 + k + 16 * half;
  bf16* la = sA + row * LDH + 16 * half;
  bf16* lb = sB + row * LDH + 16 * half;
  async_b128(la,     ga);
  async_b128(la + 8, ga + 8);
  async_b128(lb,     gb);
  async_b128(lb + 8, gb + 8);
}

static __device__ inline v16bf lds_a_frag(const bf16* sA, int row, int g) {
  const bf16* p = sA + row * LDH + 16 * g;
  return join16(*(const v8bf*)p, *(const v8bf*)(p + 8));
}

static __device__ inline void compute_step(const bf16* sA, const bf16* sB,
                                           int wid, int lm, int lg, v8f acc[2][4]) {
  v16bf a0 = lds_a_frag(sA, (wid >> 1) * 32 + lm, lg);
  v16bf a1 = lds_a_frag(sA, (wid >> 1) * 32 + 16 + lm, lg);
#pragma unroll
  for (int j = 0; j < 4; ++j) {
    const bf16* p = sB + ((wid & 1) * 64 + j * 16 + lm) * LDH + 16 * lg;
    v16bf b = join16(*(const v8bf*)p, *(const v8bf*)(p + 8));
    acc[0][j] = WMMA_BF16(a0, b, acc[0][j]);
    acc[1][j] = WMMA_BF16(a1, b, acc[1][j]);
  }
}

// Runs the whole double-buffered K loop; smem = [2 buf][2 (A,B)][TILE_HALVES]
static __device__ inline void gemm_loop(const bf16* __restrict__ Xg,
                                        const bf16* __restrict__ Wg,
                                        int m0b, int n0b,
                                        bf16 (*smem)[2][TILE_HALVES],
                                        int tid, int wid, int lm, int lg,
                                        v8f acc[2][4]) {
  issue_tiles(Xg, Wg, m0b, n0b, 0, smem[0][0], smem[0][1], tid);
  int buf = 0;
#pragma unroll 1                     // single body -> accumulators stay pinned
  for (int ks = 0; ks < 31; ++ks) {
    issue_tiles(Xg, Wg, m0b, n0b, (ks + 1) * 32, smem[buf ^ 1][0], smem[buf ^ 1][1], tid);
    wait_async_le4();                    // current buffer's 4 loads done
    __syncthreads();
    compute_step(smem[buf][0], smem[buf][1], wid, lm, lg, acc);
    __syncthreads();                     // all waves done reading before overwrite
    buf ^= 1;
  }
  wait_async_0();
  __syncthreads();
  compute_step(smem[buf][0], smem[buf][1], wid, lm, lg, acc);
}

// ---------------------------------------------------------------------------
// Kernel 0: fp32 -> bf16 convert
// ---------------------------------------------------------------------------
__global__ __launch_bounds__(256) void cvt_f32_bf16(const float* __restrict__ s,
                                                    bf16* __restrict__ d, int n) {
  int i = blockIdx.x * 256 + threadIdx.x;
  if (i < n) d[i] = (bf16)s[i];
}

// ---------------------------------------------------------------------------
// Kernel 1: QKV GEMM.  qkv = X[8192,1024] @ Wqkv^T -> scatter to
//   Q[b,h,t,d] (x 1/8), K[b,h,t,d], V[b,h,d,t] (transposed via LDS for
//   coalesced stores along t).
// ---------------------------------------------------------------------------
__global__ __launch_bounds__(256) void qkv_gemm(const bf16* __restrict__ X,
                                                const bf16* __restrict__ W,
                                                bf16* __restrict__ Qb,
                                                bf16* __restrict__ Kb,
                                                bf16* __restrict__ Vt) {
  __shared__ __align__(16) bf16 smem[2][2][TILE_HALVES];   // 40 KB
  __shared__ __align__(16) bf16 vtile[8][64 * LDH];        // 40 KB (per-wave)
  const int tid = threadIdx.x;
  const int lane = tid & 31, wid = tid >> 5;
  const int lm = lane & 15, lg = lane >> 4;
  const int m0 = blockIdx.y * 128 + (wid >> 1) * 32;
  const int n0 = blockIdx.x * 128 + (wid & 1) * 64;

  v8f acc[2][4];
#pragma unroll
  for (int i = 0; i < 2; ++i)
#pragma unroll
    for (int j = 0; j < 4; ++j) acc[i][j] = vzero8();

  gemm_loop(X, W, blockIdx.y * 128, blockIdx.x * 128, smem, tid, wid, lm, lg, acc);

  const int which = n0 >> 10;            // uniform per wave (n0 multiple of 64)
  if (which < 2) {
    // Q / K: lanes are consecutive in d -> coalesced 32B runs
#pragma unroll
    for (int mi = 0; mi < 2; ++mi)
#pragma unroll
      for (int j = 0; j < 4; ++j)
#pragma unroll
        for (int r = 0; r < 8; ++r) {
          int n = n0 + j * 16 + lm;
          int m = m0 + mi * 16 + r + 8 * lg;
          int cc = n & 1023, h = cc >> 6, d = cc & 63;
          int b = m >> 11, t = m & 2047;
          size_t bh = (size_t)(b * 16 + h);
          float v = acc[mi][j][r];
          if (which == 0)
            Qb[(bh * 2048 + t) * 64 + d] = (bf16)(v * 0.125f);  // fold 1/sqrt(64)
          else
            Kb[(bh * 2048 + t) * 64 + d] = (bf16)v;
        }
  } else {
    // V: transpose 64(d) x 32(t) through per-wave LDS, store contiguous in t
    bf16* vt = vtile[wid];
#pragma unroll
    for (int mi = 0; mi < 2; ++mi)
#pragma unroll
      for (int j = 0; j < 4; ++j)
#pragma unroll
        for (int r = 0; r < 8; ++r) {
          int n_local = j * 16 + lm;
          int m_local = mi * 16 + r + 8 * lg;
          vt[n_local * LDH + m_local] = (bf16)acc[mi][j][r];
        }
    // DS ops are in-order within a wave: safe to read back without barrier
    const int b = m0 >> 11, t0 = m0 & 2047;
#pragma unroll
    for (int cc = 0; cc < 2; ++cc) {
      int n_local = lane + 32 * cc;                 // 0..63
      int n = n0 + n_local;
      int d = n & 63, h = (n & 1023) >> 6;
      size_t bh = (size_t)(b * 16 + h);
      const bf16* src = vt + n_local * LDH;
      bf16* dst = Vt + (bh * 64 + d) * 2048 + t0;   // 64B-aligned
      *(v8bf*)(dst)      = *(const v8bf*)(src);
      *(v8bf*)(dst + 8)  = *(const v8bf*)(src + 8);
      *(v8bf*)(dst + 16) = *(const v8bf*)(src + 16);
      *(v8bf*)(dst + 24) = *(const v8bf*)(src + 24);
    }
  }
}

// ---------------------------------------------------------------------------
// Kernel 2: banded flash attention.  One wave per 16-query tile.
// S = Q K^T (scale pre-folded), online softmax (fp32), O = P V.
// Per-wave LDS tile transposes P (C/D layout) into an A fragment.
// No block barriers inside the loop (waves have different band lengths).
// ---------------------------------------------------------------------------
__global__ __launch_bounds__(256) void attn_swa(const bf16* __restrict__ Qb,
                                                const bf16* __restrict__ Kb,
                                                const bf16* __restrict__ Vt,
                                                bf16* __restrict__ Y) {
  __shared__ __align__(16) bf16 lsP[8][16 * 32];   // 1 KB per wave
  const int lane = threadIdx.x & 31, wid = threadIdx.x >> 5;
  const int lm = lane & 15, lg = lane >> 4;

  const int w  = blockIdx.x * 8 + wid;   // 8192 waves total
  const int qt = w & 127;                // T/16 = 128 tiles
  const int bh = w >> 7;                 // [0,64)
  const int q0 = qt * 16;

  const bf16* Qh = Qb + (size_t)bh * 2048 * 64;
  const bf16* Kh = Kb + (size_t)bh * 2048 * 64;
  const bf16* Vh = Vt + (size_t)bh * 64 * 2048;
  bf16* myP = lsP[wid];

  v16bf aq0 = load_a_g(Qh, 64, q0 + lm, 0, lg);
  v16bf aq1 = load_a_g(Qh, 64, q0 + lm, 32, lg);

  v8f o[4];
#pragma unroll
  for (int t = 0; t < 4; ++t) o[t] = vzero8();
  float mx[8], sm[8];
#pragma unroll
  for (int r = 0; r < 8; ++r) { mx[r] = -1e30f; sm[r] = 0.0f; }

  int kb0 = q0 - 240;                // q0 + 16 - WINDOW
  if (kb0 < 0) kb0 = 0;
  kb0 &= ~31;

#pragma unroll 1
  for (int kb = kb0; kb <= q0 + 15; kb += 32) {
    v8f s0 = vzero8(), s1 = vzero8();
#pragma unroll
    for (int c = 0; c < 2; ++c) {               // contraction over head dim
      v16bf bk0 = load_b_g(Kh + (size_t)(kb + lm) * 64 + 32 * c + 16 * lg);
      v16bf bk1 = load_b_g(Kh + (size_t)(kb + 16 + lm) * 64 + 32 * c + 16 * lg);
      v16bf ac = c ? aq1 : aq0;
      s0 = WMMA_BF16(ac, bk0, s0);
      s1 = WMMA_BF16(ac, bk1, s1);
    }

#pragma unroll
    for (int r = 0; r < 8; ++r) {
      int row = q0 + r + 8 * lg;
      int c0 = kb + lm, c1 = kb + 16 + lm;
      float v0 = (c0 <= row && c0 > row - 256) ? s0[r] : -1e30f;
      float v1 = (c1 <= row && c1 > row - 256) ? s1[r] : -1e30f;
      float mnew = fmaxf(v0, v1);
      mnew = fmaxf(mnew, __shfl_xor(mnew, 1));
      mnew = fmaxf(mnew, __shfl_xor(mnew, 2));
      mnew = fmaxf(mnew, __shfl_xor(mnew, 4));
      mnew = fmaxf(mnew, __shfl_xor(mnew, 8));
      mnew = fmaxf(mnew, mx[r]);
      float corr = __expf(mx[r] - mnew);
      mx[r] = mnew;
      float p0 = __expf(v0 - mnew), p1 = __expf(v1 - mnew);
      float rs = p0 + p1;
      rs += __shfl_xor(rs, 1);
      rs += __shfl_xor(rs, 2);
      rs += __shfl_xor(rs, 4);
      rs += __shfl_xor(rs, 8);
      sm[r] = sm[r] * corr + rs;
#pragma unroll
      for (int t = 0; t < 4; ++t) o[t][r] *= corr;
      int prow = r + 8 * lg;
      myP[prow * 32 + lm]      = (bf16)p0;       // DS in-order within wave
      myP[prow * 32 + 16 + lm] = (bf16)p1;
    }

    v16bf pA = join16(*(const v8bf*)(myP + lm * 32 + 16 * lg),
                      *(const v8bf*)(myP + lm * 32 + 8 + 16 * lg));

#pragma unroll
    for (int t = 0; t < 4; ++t) {
      v16bf vb = load_b_g(Vh + (size_t)(t * 16 + lm) * 2048 + kb + 16 * lg);
      o[t] = WMMA_BF16(pA, vb, o[t]);
    }
  }

  const int b = bh >> 4, h = bh & 15;
#pragma unroll
  for (int t = 0; t < 4; ++t)
#pragma unroll
    for (int r = 0; r < 8; ++r) {
      int row = q0 + r + 8 * lg;
      int col = h * 64 + t * 16 + lm;
      Y[((size_t)b * 2048 + row) * 1024 + col] = (bf16)(o[t][r] / sm[r]);
    }
}

// ---------------------------------------------------------------------------
// Kernel 3: output projection.  out = Y[8192,1024] @ Wproj^T, fp32 out
// ---------------------------------------------------------------------------
__global__ __launch_bounds__(256) void proj_gemm(const bf16* __restrict__ X,
                                                 const bf16* __restrict__ W,
                                                 float* __restrict__ Out) {
  __shared__ __align__(16) bf16 smem[2][2][TILE_HALVES];   // 40 KB
  const int tid = threadIdx.x;
  const int lane = tid & 31, wid = tid >> 5;
  const int lm = lane & 15, lg = lane >> 4;
  const int m0 = blockIdx.y * 128 + (wid >> 1) * 32;
  const int n0 = blockIdx.x * 128 + (wid & 1) * 64;

  v8f acc[2][4];
#pragma unroll
  for (int i = 0; i < 2; ++i)
#pragma unroll
    for (int j = 0; j < 4; ++j) acc[i][j] = vzero8();

  gemm_loop(X, W, blockIdx.y * 128, blockIdx.x * 128, smem, tid, wid, lm, lg, acc);

#pragma unroll
  for (int mi = 0; mi < 2; ++mi)
#pragma unroll
    for (int j = 0; j < 4; ++j)
#pragma unroll
      for (int r = 0; r < 8; ++r) {
        int n = n0 + j * 16 + lm;
        int m = m0 + mi * 16 + r + 8 * lg;
        Out[(size_t)m * 1024 + n] = acc[mi][j][r];
      }
}

// ---------------------------------------------------------------------------
// Host launcher
// ---------------------------------------------------------------------------
extern "C" void kernel_launch(void* const* d_in, const int* in_sizes, int n_in,
                              void* d_out, int out_size, void* d_ws, size_t ws_size,
                              hipStream_t stream) {
  const float* x     = (const float*)d_in[0];   // [4,2048,1024]
  const float* wqkv  = (const float*)d_in[1];   // [3072,1024]
  const float* wproj = (const float*)d_in[2];   // [1024,1024]
  float* out = (float*)d_out;                   // [4,2048,1024] fp32

  bf16* xb     = (bf16*)d_ws;                       // 8192*1024
  bf16* wqkvb  = xb + (size_t)8192 * 1024;          // 3072*1024
  bf16* wprojb = wqkvb + (size_t)3072 * 1024;       // 1024*1024
  bf16* Qb     = wprojb + (size_t)1024 * 1024;      // 64*2048*64
  bf16* Kb     = Qb + (size_t)64 * 2048 * 64;       // 64*2048*64
  bf16* Vtb    = Kb + (size_t)64 * 2048 * 64;       // 64*64*2048 (transposed)
  bf16* ybuf   = Vtb + (size_t)64 * 64 * 2048;      // 8192*1024

  (void)in_sizes; (void)n_in; (void)out_size; (void)ws_size;

  const int nX = 8192 * 1024, nWq = 3072 * 1024, nWp = 1024 * 1024;
  cvt_f32_bf16<<<(nX + 255) / 256, 256, 0, stream>>>(x, xb, nX);
  cvt_f32_bf16<<<(nWq + 255) / 256, 256, 0, stream>>>(wqkv, wqkvb, nWq);
  cvt_f32_bf16<<<(nWp + 255) / 256, 256, 0, stream>>>(wproj, wprojb, nWp);

  qkv_gemm<<<dim3(24, 64), 256, 0, stream>>>(xb, wqkvb, Qb, Kb, Vtb);
  attn_swa<<<1024, 256, 0, stream>>>(Qb, Kb, Vtb, ybuf);
  proj_gemm<<<dim3(8, 64), 256, 0, stream>>>(ybuf, wprojb, out);
}
